// ODEBlockG_3908420240051
// MI455X (gfx1250) — matrix operand order repaired and checked by reading, not compiled, not run on hardware
//
#include <hip/hip_runtime.h>
#include <hip/hip_bf16.h>

typedef __attribute__((ext_vector_type(16))) _Float16 v16h;
typedef __attribute__((ext_vector_type(8)))  float    v8f;
typedef __attribute__((ext_vector_type(8)))  _Float16 h8v;
typedef __attribute__((ext_vector_type(8)))  float    f8v;

#define D_IN   256
#define D_HID  512
#define BATCH  65536
#define NSTEP  8

// Hardware tanh (v_tanh_f32) when the builtin exists on this toolchain.
__device__ __forceinline__ float fast_tanh(float x) {
#if __has_builtin(__builtin_amdgcn_tanhf)
  return __builtin_amdgcn_tanhf(x);
#else
  return tanhf(x);
#endif
}

// ---------------------------------------------------------------------------
// Weight transpose + fp32 -> f16 convert:  W[K,N] (row major) -> WT[N,K] f16
// ---------------------------------------------------------------------------
__global__ void transpose_f16_kernel(const float* __restrict__ W,
                                     _Float16* __restrict__ WT,
                                     int K, int N) {
  int i = blockIdx.x * blockDim.x + threadIdx.x;
  if (i < K * N) {
    int k = i / N;
    int n = i - k * N;
    WT[(size_t)n * K + k] = (_Float16)W[i];
  }
}

// ---------------------------------------------------------------------------
// y = x ; s16 = f16(x)        (8 elements / thread, 32B loads)
// ---------------------------------------------------------------------------
__global__ void init_kernel(const float* __restrict__ x,
                            float* __restrict__ y,
                            _Float16* __restrict__ s16, int n8) {
  int i = blockIdx.x * blockDim.x + threadIdx.x;
  if (i >= n8) return;
  size_t o = (size_t)i * 8;
  f8v v = *(const f8v*)(x + o);
  *(f8v*)(y + o) = v;
  *(h8v*)(s16 + o) = __builtin_convertvector(v, h8v);
}

// ---------------------------------------------------------------------------
// out16 = f16( y + c1*k1 + ... + c<cnt>*k<cnt> )     (RK stage input)
// ---------------------------------------------------------------------------
__global__ void combine_kernel(const float* __restrict__ y,
                               const _Float16* __restrict__ k1,
                               const _Float16* __restrict__ k2,
                               const _Float16* __restrict__ k3,
                               const _Float16* __restrict__ k4,
                               const _Float16* __restrict__ k5,
                               float c1, float c2, float c3, float c4, float c5,
                               int cnt,
                               _Float16* __restrict__ out, int n8) {
  int i = blockIdx.x * blockDim.x + threadIdx.x;
  if (i >= n8) return;
  size_t o = (size_t)i * 8;
  f8v s = *(const f8v*)(y + o);
  s += c1 * __builtin_convertvector(*(const h8v*)(k1 + o), f8v);
  if (cnt > 1) s += c2 * __builtin_convertvector(*(const h8v*)(k2 + o), f8v);
  if (cnt > 2) s += c3 * __builtin_convertvector(*(const h8v*)(k3 + o), f8v);
  if (cnt > 3) s += c4 * __builtin_convertvector(*(const h8v*)(k4 + o), f8v);
  if (cnt > 4) s += c5 * __builtin_convertvector(*(const h8v*)(k5 + o), f8v);
  *(h8v*)(out + o) = __builtin_convertvector(s, h8v);
}

// ---------------------------------------------------------------------------
// yout = y + c1*k1 + c3*k3 + c4*k4 + c5*k5 + c6*k6 (fp32); sout = f16(yout)
// ---------------------------------------------------------------------------
__global__ void update_kernel(const float* __restrict__ y,
                              const _Float16* __restrict__ k1,
                              const _Float16* __restrict__ k3,
                              const _Float16* __restrict__ k4,
                              const _Float16* __restrict__ k5,
                              const _Float16* __restrict__ k6,
                              float c1, float c3, float c4, float c5, float c6,
                              float* __restrict__ yout,
                              _Float16* __restrict__ sout, int n8) {
  int i = blockIdx.x * blockDim.x + threadIdx.x;
  if (i >= n8) return;
  size_t o = (size_t)i * 8;
  f8v s = *(const f8v*)(y + o);
  s += c1 * __builtin_convertvector(*(const h8v*)(k1 + o), f8v);
  s += c3 * __builtin_convertvector(*(const h8v*)(k3 + o), f8v);
  s += c4 * __builtin_convertvector(*(const h8v*)(k4 + o), f8v);
  s += c5 * __builtin_convertvector(*(const h8v*)(k5 + o), f8v);
  s += c6 * __builtin_convertvector(*(const h8v*)(k6 + o), f8v);
  *(f8v*)(yout + o) = s;
  *(h8v*)(sout + o) = __builtin_convertvector(s, h8v);
}

// ---------------------------------------------------------------------------
// GEMM:  C[BATCH,N] (f16) = act( A[BATCH,K] (f16) @ BT[N,K]^T (f16) + bias )
// fp32 accumulation via v_wmma_f32_16x16x32_f16.
// 256 threads = 8 waves; block tile 128x128; wave tile 32x64 (2x4 frags).
// Output staged through LDS for coalesced global_store_b128.
// ---------------------------------------------------------------------------
template <int N, int K, bool TANH>
__global__ __launch_bounds__(256)
void gemm_bias_act(const _Float16* __restrict__ A,
                   const _Float16* __restrict__ BT,
                   const float* __restrict__ bias,
                   _Float16* __restrict__ C) {
  constexpr int LDA = 40;   // 32 + 8 halves pad: 16B-aligned rows, banks spread
  constexpr int LDC = 136;  // 128 + 8 halves pad: 4-word bank rotation per row
  __shared__ __align__(16) _Float16 lds_a[128 * LDA];
  __shared__ __align__(16) _Float16 lds_b[128 * LDA];
  __shared__ __align__(16) _Float16 lds_c[128 * LDC];

  const int tid  = threadIdx.x;
  const int wave = tid >> 5;
  const int lane = tid & 31;
  const int l16  = lane & 15;
  const int hi   = lane >> 4;

  const int bm = blockIdx.x * 128;
  const int bn = blockIdx.y * 128;
  const int wm = (wave & 3) * 32;   // wave row offset in tile
  const int wn = (wave >> 2) * 64;  // wave col offset in tile

  v8f acc[2][4] = {};

  const int lrow = tid >> 2;        // 0..63 (two passes cover 128 rows)
  const int lcol = (tid & 3) * 8;   // half offset: 0,8,16,24

  for (int k0 = 0; k0 < K; k0 += 32) {
#pragma unroll
    for (int p = 0; p < 2; ++p) {
      const int r = lrow + p * 64;
      *(float4*)(lds_a + r * LDA + lcol) =
          *(const float4*)(A + (size_t)(bm + r) * K + (k0 + lcol));
      *(float4*)(lds_b + r * LDA + lcol) =
          *(const float4*)(BT + (size_t)(bn + r) * K + (k0 + lcol));
    }
    __syncthreads();

    // A fragments: lane l16 = row; halves 0..7 -> K=hi*8.., 8..15 -> K=16+hi*8..
    v16h af[2];
#pragma unroll
    for (int mf = 0; mf < 2; ++mf) {
      const _Float16* p = lds_a + (wm + mf * 16 + l16) * LDA + hi * 8;
      union { v16h v; float4 f[2]; } u;
      u.f[0] = *(const float4*)(p);
      u.f[1] = *(const float4*)(p + 16);
      af[mf] = u.v;
    }
    // B fragments: lane l16 = output column; halves 0..7 -> K=hi*8.., 8..15 -> K=16+hi*8..
    v16h bf[4];
#pragma unroll
    for (int nf = 0; nf < 4; ++nf) {
      const _Float16* p = lds_b + (wn + nf * 16 + l16) * LDA + hi * 8;
      union { v16h v; float4 f[2]; } u;
      u.f[0] = *(const float4*)(p);
      u.f[1] = *(const float4*)(p + 16);
      bf[nf] = u.v;
    }

#pragma unroll
    for (int mf = 0; mf < 2; ++mf)
#pragma unroll
      for (int nf = 0; nf < 4; ++nf)
        acc[mf][nf] = __builtin_amdgcn_wmma_f32_16x16x32_f16(
            false, af[mf], false, bf[nf], (short)0, acc[mf][nf], false, false);

    __syncthreads();
  }

  // Epilogue: bias (+tanh), stage f16 tile in LDS.
  // C/D layout -> VGPR r: M = r + hi*8, N = l16
#pragma unroll
  for (int nf = 0; nf < 4; ++nf) {
    const int lcol_c = wn + nf * 16 + l16;
    const float bv = bias[bn + lcol_c];
#pragma unroll
    for (int mf = 0; mf < 2; ++mf) {
      const int lrow_c = wm + mf * 16 + hi * 8;
#pragma unroll
      for (int r = 0; r < 8; ++r) {
        float v = acc[mf][nf][r] + bv;
        if (TANH) v = fast_tanh(v);
        lds_c[(lrow_c + r) * LDC + lcol_c] = (_Float16)v;
      }
    }
  }
  __syncthreads();

  // Cooperative coalesced stores: each thread writes 128B of one row.
  {
    const int row = tid >> 1;        // 0..127
    const int ch  = (tid & 1) * 64;  // half offset within row
    const size_t gbase = (size_t)(bm + row) * N + bn + ch;
    const _Float16* src = lds_c + row * LDC + ch;
#pragma unroll
    for (int j = 0; j < 8; ++j) {
      *(float4*)(C + gbase + j * 8) = *(const float4*)(src + j * 8);
    }
  }
}

// ---------------------------------------------------------------------------
extern "C" void kernel_launch(void* const* d_in, const int* in_sizes, int n_in,
                              void* d_out, int out_size, void* d_ws, size_t ws_size,
                              hipStream_t stream) {
  const float* x  = (const float*)d_in[0];
  const float* W1 = (const float*)d_in[1];
  const float* b1 = (const float*)d_in[2];
  const float* W2 = (const float*)d_in[3];
  const float* b2 = (const float*)d_in[4];

  char* ws = (char*)d_ws;
  size_t off = 0;
  auto take = [&](size_t bytes) -> void* {
    void* p = ws + off;
    off = (off + bytes + 255) & ~(size_t)255;
    return p;
  };
  _Float16* w1t = (_Float16*)take((size_t)D_HID * D_IN * 2);   // [512,256]
  _Float16* w2t = (_Float16*)take((size_t)D_IN * D_HID * 2);   // [256,512]
  float*    y   = (float*)take((size_t)BATCH * D_IN * 4);
  _Float16* s16 = (_Float16*)take((size_t)BATCH * D_IN * 2);   // stage input (f16)
  _Float16* h16 = (_Float16*)take((size_t)BATCH * D_HID * 2);  // hidden activations
  _Float16* kk[6];
  for (int i = 0; i < 6; ++i) kk[i] = (_Float16*)take((size_t)BATCH * D_IN * 2);
  (void)ws_size; (void)in_sizes; (void)n_in; (void)out_size;

  const int n8 = BATCH * D_IN / 8;   // 8 elements per thread
  const int eb = 256;
  const int eg = n8 / eb;

  transpose_f16_kernel<<<(D_IN * D_HID + 255) / 256, 256, 0, stream>>>(W1, w1t, D_IN, D_HID);
  transpose_f16_kernel<<<(D_IN * D_HID + 255) / 256, 256, 0, stream>>>(W2, w2t, D_HID, D_IN);
  init_kernel<<<eg, eb, 0, stream>>>(x, y, s16, n8);

  const dim3 g1(BATCH / 128, D_HID / 128);
  const dim3 g2(BATCH / 128, D_IN / 128);

  auto feval = [&](_Float16* kout) {
    gemm_bias_act<D_HID, D_IN, true ><<<g1, 256, 0, stream>>>(s16, w1t, b1, h16);
    gemm_bias_act<D_IN, D_HID, false><<<g2, 256, 0, stream>>>(h16, w2t, b2, kout);
  };

  // Dormand-Prince 5(4) tableau, fixed step h = 1/NSTEP
  const float h = 1.0f / (float)NSTEP;
  const float a21 = 0.2f;
  const float a31 = 3.f/40.f,        a32 = 9.f/40.f;
  const float a41 = 44.f/45.f,       a42 = -56.f/15.f,     a43 = 32.f/9.f;
  const float a51 = 19372.f/6561.f,  a52 = -25360.f/2187.f, a53 = 64448.f/6561.f, a54 = -212.f/729.f;
  const float a61 = 9017.f/3168.f,   a62 = -355.f/33.f,     a63 = 46732.f/5247.f, a64 = 49.f/176.f, a65 = -5103.f/18656.f;
  const float c1  = 35.f/384.f,      c3  = 500.f/1113.f,    c4  = 125.f/192.f,    c5 = -2187.f/6784.f, c6 = 11.f/84.f;

  for (int step = 0; step < NSTEP; ++step) {
    // k1 = f(y); s16 already holds f16(y)
    feval(kk[0]);
    combine_kernel<<<eg, eb, 0, stream>>>(y, kk[0], kk[0], kk[0], kk[0], kk[0],
        h*a21, 0.f, 0.f, 0.f, 0.f, 1, s16, n8);
    feval(kk[1]);
    combine_kernel<<<eg, eb, 0, stream>>>(y, kk[0], kk[1], kk[1], kk[1], kk[1],
        h*a31, h*a32, 0.f, 0.f, 0.f, 2, s16, n8);
    feval(kk[2]);
    combine_kernel<<<eg, eb, 0, stream>>>(y, kk[0], kk[1], kk[2], kk[2], kk[2],
        h*a41, h*a42, h*a43, 0.f, 0.f, 3, s16, n8);
    feval(kk[3]);
    combine_kernel<<<eg, eb, 0, stream>>>(y, kk[0], kk[1], kk[2], kk[3], kk[3],
        h*a51, h*a52, h*a53, h*a54, 0.f, 4, s16, n8);
    feval(kk[4]);
    combine_kernel<<<eg, eb, 0, stream>>>(y, kk[0], kk[1], kk[2], kk[3], kk[4],
        h*a61, h*a62, h*a63, h*a64, h*a65, 5, s16, n8);
    feval(kk[5]);

    float* yout = (step == NSTEP - 1) ? (float*)d_out : y;
    update_kernel<<<eg, eb, 0, stream>>>(y, kk[0], kk[2], kk[3], kk[4], kk[5],
        h*c1, h*c3, h*c4, h*c5, h*c6, yout, s16, n8);
  }
}
